// QMIL_36335423324487
// MI455X (gfx1250) — matrix operand order repaired and checked by reading, not compiled, not run on hardware
//
#include <hip/hip_runtime.h>
#include <hip/hip_bf16.h>

// ---------------------------------------------------------------------------
// MIL attention model on MI455X (gfx1250).  All heavy matmuls via
// v_wmma_f32_16x16x32_f16 (wave32, f32 accumulate).  GEMM A-tiles staged
// through LDS with global_load_async_to_lds_b128 double-buffering
// (ASYNCcnt).  LN/softmax/GELU in f32 VALU.
// ---------------------------------------------------------------------------

#define Bn    4
#define Nn    2048
#define Dm    512
#define Hh    8
#define HdD   64
#define Sreal 2049
#define Spad  2112          // multiple of 64 (GEMM M tile) and 32 (PV WMMA K)
#define NSEQ  8             // 4 neg + 4 pos sequences
#define MT    (Spad / 16)   // 132 16-row tiles
#define MT64  (Spad / 64)   // 33  64-row tiles

typedef __attribute__((ext_vector_type(16))) _Float16 v16h;
typedef __attribute__((ext_vector_type(8)))  _Float16 h8;
typedef __attribute__((ext_vector_type(8)))  float    v8f;

union U16h { v16h v; h8 h[2]; };

__device__ __forceinline__ float gelu_f(float x) {
    return 0.5f * x * (1.f + erff(x * 0.70710678118654752f));
}

__device__ __forceinline__ v8f wmma_f16(v16h a, v16h b, v8f c) {
    return __builtin_amdgcn_wmma_f32_16x16x32_f16(false, a, false, b,
                                                  (short)0, c, false, false);
}

// Async global->LDS copy, ASYNCcnt-tracked (CDNA5 ISA).
__device__ __forceinline__ void async_copy_b128(unsigned lds_off,
                                                const void* gptr) {
    asm volatile("global_load_async_to_lds_b128 %0, %1, off"
                 :: "v"(lds_off), "v"((unsigned long long)(size_t)gptr)
                 : "memory");
}
__device__ __forceinline__ void async_wait(int pending) {
    if (pending) asm volatile("s_wait_asynccnt 8" ::: "memory");
    else         asm volatile("s_wait_asynccnt 0" ::: "memory");
}

// ---------------------------------------------------------------------------
// f32 -> f16 weight conversion
// ---------------------------------------------------------------------------
__global__ void cvt_f32_f16(const float* __restrict__ s,
                            _Float16* __restrict__ d, int n) {
    int i = blockIdx.x * 256 + threadIdx.x;
    if (i < n) d[i] = (_Float16)s[i];
}

// ---------------------------------------------------------------------------
// cls row + zero padded rows + their masks.  grid (64, NSEQ), block 256.
// slot 0 -> row 0 (cls, mask 1); slot 1..63 -> rows 2049..2111 (zero, mask 0)
// ---------------------------------------------------------------------------
__global__ __launch_bounds__(256)
void init_kernel(const float* __restrict__ cls,
                 _Float16* __restrict__ seqbuf, float* __restrict__ maskbuf) {
    int slot = blockIdx.x, s = blockIdx.y, t = threadIdx.x;
    int row = (slot == 0) ? 0 : (Nn + slot);
    _Float16* d = seqbuf + ((long)s * Spad + row) * Dm;
    if (slot == 0) {
        d[t]       = (_Float16)cls[t];
        d[t + 256] = (_Float16)cls[t + 256];
        if (t == 0) maskbuf[(long)s * Spad + row] = 1.f;
    } else {
        d[t]       = (_Float16)0.f;
        d[t + 256] = (_Float16)0.f;
        if (t == 0) maskbuf[(long)s * Spad + row] = 0.f;
    }
}

// ---------------------------------------------------------------------------
// Per-instance attention weight MLP (D->64->64->1, gelu, sigmoid) and the two
// scaled embedding rows.  One 64-thread block per instance.
// ---------------------------------------------------------------------------
__global__ __launch_bounds__(64)
void aw_kernel(const float* __restrict__ x,
               const float* __restrict__ W0, const float* __restrict__ b0,
               const float* __restrict__ W1, const float* __restrict__ b1,
               const float* __restrict__ W2, const float* __restrict__ b2,
               _Float16* __restrict__ seqbuf, float* __restrict__ maskbuf) {
    __shared__ float xs[Dm], h0s[64], h1s[64], red[64];
    int inst = blockIdx.x;
    int b = inst / Nn, n = inst % Nn, t = threadIdx.x;
    const float* xr = x + (long)inst * Dm;
    for (int i = t; i < Dm; i += 64) xs[i] = xr[i];
    __syncthreads();
    float acc = b0[t];
    for (int k = 0; k < Dm; ++k) acc += xs[k] * W0[(long)t * Dm + k];
    h0s[t] = gelu_f(acc);
    __syncthreads();
    acc = b1[t];
    for (int k = 0; k < 64; ++k) acc += h0s[k] * W1[t * 64 + k];
    h1s[t] = gelu_f(acc);
    __syncthreads();
    red[t] = h1s[t] * W2[t];
    __syncthreads();
    for (int o = 32; o >= 1; o >>= 1) {
        if (t < o) red[t] += red[t + o];
        __syncthreads();
    }
    float w = 1.f / (1.f + __expf(-(red[0] + b2[0])));
    if (t == 0) {
        maskbuf[(long)b * Spad + 1 + n]       = (w < 0.5f) ? 1.f : 0.f;
        maskbuf[(long)(4 + b) * Spad + 1 + n] = (w < 0.5f) ? 0.f : 1.f;
    }
    float wn = w * sqrtf(w);                 // w^1.5
    float wp = powf(w, 2.0f / 3.0f);         // w^(2/3)
    _Float16* d0 = seqbuf + ((long)b * Spad + 1 + n) * Dm;
    _Float16* d1 = seqbuf + ((long)(4 + b) * Spad + 1 + n) * Dm;
    for (int i = t; i < Dm; i += 64) {
        d0[i] = (_Float16)(xs[i] * wn);
        d1[i] = (_Float16)(xs[i] * wp);
    }
}

// ---------------------------------------------------------------------------
// LayerNorm over D=512: out = ln(xin [+ add]) * g + b.  grid (Spad, NSEQ).
// ---------------------------------------------------------------------------
__global__ __launch_bounds__(256)
void ln_kernel(const _Float16* __restrict__ xin,
               const float* __restrict__ add,          // nullable
               const float* __restrict__ gv, const float* __restrict__ bv,
               _Float16* __restrict__ out) {
    __shared__ float red[256];
    long r = (long)blockIdx.y * Spad + blockIdx.x;
    int t = threadIdx.x;
    const _Float16* xp = xin + r * Dm;
    float v0 = (float)xp[t], v1 = (float)xp[t + 256];
    if (add) { v0 += add[r * Dm + t]; v1 += add[r * Dm + t + 256]; }
    red[t] = v0 + v1;
    __syncthreads();
    for (int o = 128; o > 0; o >>= 1) {
        if (t < o) red[t] += red[t + o];
        __syncthreads();
    }
    float mean = red[0] * (1.f / Dm);
    __syncthreads();
    float d0 = v0 - mean, d1 = v1 - mean;
    red[t] = d0 * d0 + d1 * d1;
    __syncthreads();
    for (int o = 128; o > 0; o >>= 1) {
        if (t < o) red[t] += red[t + o];
        __syncthreads();
    }
    float rs = rsqrtf(red[0] * (1.f / Dm) + 1e-5f);
    out[r * Dm + t]       = (_Float16)(d0 * rs * gv[t] + bv[t]);
    out[r * Dm + t + 256] = (_Float16)(d1 * rs * gv[t + 256] + bv[t + 256]);
}

// ---------------------------------------------------------------------------
// WMMA GEMM: Out[M,N] = A[M,K] @ W[N,K]^T + bias.  One wave per 64x64 tile
// (16 accumulators), K in steps of 32.  A staged via async-to-LDS ping-pong:
// 8 global_load_async_to_lds_b128 per k-step (64 rows x 32 halves), wait
// s_wait_asynccnt 8 so the previous stage is resident (in-order completion),
// compute 16 WMMAs per k-step (4 per B load).
//   A layout: lane m=ln, a[i] = A[m][k0 + (i>>3)*16 + g*8 + (i&7)]
//   B layout: lane n=ln, b[i] = W[n][k0 + (i>>3)*16 + g*8 + (i&7)]
//   C layout: c[j]      = C[g*8 + j][n]
// ---------------------------------------------------------------------------
template <bool OUT_F16>
__global__ __launch_bounds__(32)
void gemm_wmma(const _Float16* __restrict__ A, int lda, long aSeq,
               const _Float16* __restrict__ W,
               const float* __restrict__ bias,
               void* __restrict__ Out, int ldo, long oSeq, int K) {
    __shared__ __align__(16) _Float16 atile[2][64 * 32];   // 2 x 4KB stages
    int lane = threadIdx.x, g = lane >> 4, ln = lane & 15;
    int m0 = blockIdx.x * 64, n0 = blockIdx.y * 64, s = blockIdx.z;
    const _Float16* Abase = A + (long)s * aSeq + (long)m0 * lda;
    unsigned ldsb[2] = { (unsigned)(size_t)&atile[0][0],
                         (unsigned)(size_t)&atile[1][0] };

    // stage one k-step of A (64 rows x 32 halves) with 8 async b128 copies:
    // instruction q: row = (q&3)*16 + ln, 8-half segment = (q>>2)*2 + g
    auto issueA = [&](int kstep, int st) {
#pragma unroll
        for (int q = 0; q < 8; ++q) {
            int row = (q & 3) * 16 + ln;
            int seg = (q >> 2) * 2 + g;                    // 0..3
            const _Float16* gp = Abase + (long)row * lda + kstep * 32 + seg * 8;
            async_copy_b128(ldsb[st] + (unsigned)(row * 64 + seg * 16), gp);
        }
    };

    v8f c[4][4] = {};
    int steps = K >> 5;
    issueA(0, 0);
    for (int i = 0; i < steps; ++i) {
        if (i + 1 < steps) { issueA(i + 1, (i + 1) & 1); async_wait(1); }
        else               { async_wait(0); }
        int st = i & 1;
        U16h a[4];
#pragma unroll
        for (int mt = 0; mt < 4; ++mt) {
            a[mt].h[0] = *(const h8*)&atile[st][(mt * 16 + ln) * 32 + g * 8];
            a[mt].h[1] = *(const h8*)&atile[st][(mt * 16 + ln) * 32 + 16 + g * 8];
        }
#pragma unroll
        for (int nt = 0; nt < 4; ++nt) {
            const _Float16* Wr = W + (long)(n0 + nt * 16 + ln) * K + i * 32 + g * 8;
            U16h b;
            b.h[0] = *(const h8*)(Wr);
            b.h[1] = *(const h8*)(Wr + 16);
#pragma unroll
            for (int mt = 0; mt < 4; ++mt)
                c[mt][nt] = wmma_f16(a[mt].v, b.v, c[mt][nt]);
        }
    }
#pragma unroll
    for (int nt = 0; nt < 4; ++nt) {
        int col = n0 + nt * 16 + ln;
        float bb = bias[col];
#pragma unroll
        for (int mt = 0; mt < 4; ++mt)
#pragma unroll
            for (int j = 0; j < 8; ++j) {
                long idx = (long)s * oSeq +
                           (long)(m0 + mt * 16 + g * 8 + j) * ldo + col;
                float v = c[mt][nt][j] + bb;
                if (OUT_F16) ((_Float16*)Out)[idx] = (_Float16)v;
                else         ((float*)Out)[idx] = v;
            }
    }
}

// ---------------------------------------------------------------------------
// Flash attention over one (seq, head, 16-query-row tile) per wave.
// qkv: [NSEQ][Spad][1536] f16 (Q | K | V per row), mask: [NSEQ][Spad] f32,
// obuf: [NSEQ][Spad][512] f16 (heads concatenated at col h*64).
// ---------------------------------------------------------------------------
__global__ __launch_bounds__(32)
void attn_wmma(const _Float16* __restrict__ qkv,
               const float* __restrict__ mask,
               _Float16* __restrict__ obuf) {
    __shared__ __align__(16) _Float16 pbuf[16 * 32];   // P tile, C->A transpose
    int lane = threadIdx.x, g = lane >> 4, ln = lane & 15;
    int qt = blockIdx.x, h = blockIdx.y, s = blockIdx.z;
    long base = (long)s * Spad * 1536;
    const _Float16* Q = qkv + base + (long)(qt * 16 + ln) * 1536 + h * 64;
    U16h aq0, aq1;
    aq0.h[0] = *(const h8*)(Q + g * 8);
    aq0.h[1] = *(const h8*)(Q + 16 + g * 8);
    aq1.h[0] = *(const h8*)(Q + 32 + g * 8);
    aq1.h[1] = *(const h8*)(Q + 48 + g * 8);

    float mrun[8], lrun[8];
#pragma unroll
    for (int j = 0; j < 8; ++j) { mrun[j] = -1e30f; lrun[j] = 0.f; }
    v8f o[4] = {};
    const float scale = 0.125f;                         // 1/sqrt(64)

    for (int kb = 0; kb < Spad; kb += 32) {
        v8f sc[2];
#pragma unroll
        for (int t = 0; t < 2; ++t) {
            int key = kb + t * 16 + ln;
            const _Float16* Kr = qkv + base + (long)key * 1536 + 512 + h * 64;
            U16h b0, b1;
            b0.h[0] = *(const h8*)(Kr + g * 8);
            b0.h[1] = *(const h8*)(Kr + 16 + g * 8);
            b1.h[0] = *(const h8*)(Kr + 32 + g * 8);
            b1.h[1] = *(const h8*)(Kr + 48 + g * 8);
            v8f cc = {};
            cc = wmma_f16(aq0.v, b0.v, cc);
            cc = wmma_f16(aq1.v, b1.v, cc);
            float mv = mask[(long)s * Spad + key];
#pragma unroll
            for (int j = 0; j < 8; ++j)
                cc[j] = (mv > 0.5f) ? cc[j] * scale : -1e9f;
            sc[t] = cc;
        }
        // online softmax per output row r = g*8+j (16-lane reductions)
#pragma unroll
        for (int j = 0; j < 8; ++j) {
            float rm = fmaxf(sc[0][j], sc[1][j]);
            rm = fmaxf(rm, __shfl_xor(rm, 1, 32));
            rm = fmaxf(rm, __shfl_xor(rm, 2, 32));
            rm = fmaxf(rm, __shfl_xor(rm, 4, 32));
            rm = fmaxf(rm, __shfl_xor(rm, 8, 32));
            float mnew = fmaxf(mrun[j], rm);
            float f = __expf(mrun[j] - mnew);
            mrun[j] = mnew;
            float p0 = __expf(sc[0][j] - mnew);
            float p1 = __expf(sc[1][j] - mnew);
            float ps = p0 + p1;
            ps += __shfl_xor(ps, 1, 32);
            ps += __shfl_xor(ps, 2, 32);
            ps += __shfl_xor(ps, 4, 32);
            ps += __shfl_xor(ps, 8, 32);
            lrun[j] = lrun[j] * f + ps;
            o[0][j] *= f; o[1][j] *= f; o[2][j] *= f; o[3][j] *= f;
            int row = g * 8 + j;
            pbuf[row * 32 + ln]      = (_Float16)p0;
            pbuf[row * 32 + 16 + ln] = (_Float16)p1;
        }
        // wave-internal LDS transpose: DS ops are in-order; fence explicitly
        asm volatile("s_wait_dscnt 0" ::: "memory");
        U16h ap;
        ap.h[0] = *(const h8*)(&pbuf[ln * 32 + g * 16]);
        ap.h[1] = *(const h8*)(&pbuf[ln * 32 + g * 16 + 8]);
        asm volatile("" ::: "memory");
        // O += P @ V   (A = P 16x32, B = V[32 keys x 16 cols] per n-tile)
#pragma unroll
        for (int nt = 0; nt < 4; ++nt) {
            U16h bv;
            const _Float16* Vb = qkv + base + (long)(kb + g * 16) * 1536 +
                                 1024 + h * 64 + nt * 16 + ln;
#pragma unroll
            for (int i = 0; i < 16; ++i) bv.v[i] = Vb[(long)i * 1536];
            o[nt] = wmma_f16(ap.v, bv.v, o[nt]);
        }
    }
#pragma unroll
    for (int nt = 0; nt < 4; ++nt)
#pragma unroll
        for (int j = 0; j < 8; ++j) {
            float v = o[nt][j] / lrun[j];
            obuf[(long)s * Spad * Dm + (long)(qt * 16 + g * 8 + j) * Dm +
                 h * 64 + nt * 16 + ln] = (_Float16)v;
        }
}

// ---------------------------------------------------------------------------
// Final projection MLP: [neg_cls | pos_cls] (1024) -> 512 -> 512 -> 512.
// One 256-thread block per batch element.
// ---------------------------------------------------------------------------
__global__ __launch_bounds__(256)
void proj_kernel(const _Float16* __restrict__ abuf,
                 const float* __restrict__ W0, const float* __restrict__ b0,
                 const float* __restrict__ W1, const float* __restrict__ b1,
                 const float* __restrict__ W2, const float* __restrict__ b2,
                 float* __restrict__ out) {
    __shared__ float in[1024], hbuf[512];
    int b = blockIdx.x, t = threadIdx.x;
    const _Float16* ng = abuf + (long)b * Spad * Dm;        // row 0 = cls
    const _Float16* ps = abuf + (long)(4 + b) * Spad * Dm;
    in[t] = (float)ng[t];           in[t + 256] = (float)ng[t + 256];
    in[512 + t] = (float)ps[t];     in[768 + t] = (float)ps[t + 256];
    __syncthreads();
    float a0 = b0[t], a1 = b0[t + 256];
    for (int k = 0; k < 1024; ++k) {
        a0 += in[k] * W0[(long)t * 1024 + k];
        a1 += in[k] * W0[(long)(t + 256) * 1024 + k];
    }
    hbuf[t] = gelu_f(a0); hbuf[t + 256] = gelu_f(a1);
    __syncthreads();
    a0 = b1[t]; a1 = b1[t + 256];
    for (int k = 0; k < 512; ++k) {
        a0 += hbuf[k] * W1[(long)t * 512 + k];
        a1 += hbuf[k] * W1[(long)(t + 256) * 512 + k];
    }
    in[t] = gelu_f(a0); in[t + 256] = gelu_f(a1);
    __syncthreads();
    a0 = b2[t]; a1 = b2[t + 256];
    for (int k = 0; k < 512; ++k) {
        a0 += in[k] * W2[(long)t * 512 + k];
        a1 += in[k] * W2[(long)(t + 256) * 512 + k];
    }
    out[(long)b * 512 + t] = a0;
    out[(long)b * 512 + t + 256] = a1;
}

// ---------------------------------------------------------------------------
extern "C" void kernel_launch(void* const* d_in, const int* in_sizes, int n_in,
                              void* d_out, int out_size, void* d_ws,
                              size_t ws_size, hipStream_t stream) {
    const float* x       = (const float*)d_in[0];
    const float* cls     = (const float*)d_in[1];
    const float* ln_g    = (const float*)d_in[2];
    const float* ln_b    = (const float*)d_in[3];
    const float* aw_W0   = (const float*)d_in[4];
    const float* aw_b0   = (const float*)d_in[5];
    const float* aw_W1   = (const float*)d_in[6];
    const float* aw_b1   = (const float*)d_in[7];
    const float* aw_W2   = (const float*)d_in[8];
    const float* aw_b2   = (const float*)d_in[9];
    const float* pr_W0   = (const float*)d_in[10];
    const float* pr_b0   = (const float*)d_in[11];
    const float* pr_W1   = (const float*)d_in[12];
    const float* pr_b1   = (const float*)d_in[13];
    const float* pr_W2   = (const float*)d_in[14];
    const float* pr_b2   = (const float*)d_in[15];
    const float* Wqkv_f32[2] = {(const float*)d_in[16], (const float*)d_in[20]};
    const float* bqkv[2]     = {(const float*)d_in[17], (const float*)d_in[21]};
    const float* Wo_f32[2]   = {(const float*)d_in[18], (const float*)d_in[22]};
    const float* bo[2]       = {(const float*)d_in[19], (const float*)d_in[23]};
    float* out = (float*)d_out;

    // ---- workspace carve-up (~126 MB) ----
    char* wsp = (char*)d_ws;
    auto carve = [&](size_t bytes) {
        char* r = wsp;
        wsp += (bytes + 255) & ~(size_t)255;
        return r;
    };
    const long SZ_WQKV = 1536L * 512, SZ_WO = 512L * 512;
    const long H_SEQ = (long)NSEQ * Spad * Dm;       // halves
    const long H_QKV = (long)NSEQ * Spad * 1536;     // halves
    _Float16* wqkv16[2], *wo16[2];
    wqkv16[0] = (_Float16*)carve(2 * SZ_WQKV);
    wo16[0]   = (_Float16*)carve(2 * SZ_WO);
    wqkv16[1] = (_Float16*)carve(2 * SZ_WQKV);
    wo16[1]   = (_Float16*)carve(2 * SZ_WO);
    _Float16* seqbuf = (_Float16*)carve(2 * H_SEQ);
    _Float16* zbuf   = (_Float16*)carve(2 * H_SEQ);
    _Float16* abuf   = (_Float16*)carve(2 * H_SEQ);
    _Float16* obuf   = (_Float16*)carve(2 * H_SEQ);
    _Float16* qkvbuf = (_Float16*)carve(2 * H_QKV);
    float*    mhabuf = (float*)qkvbuf;               // alias: qkv dead by then
    float*    maskbuf = (float*)carve(4L * NSEQ * Spad);

    // ---- weights f32 -> f16 ----
    for (int l = 0; l < 2; ++l) {
        cvt_f32_f16<<<(int)((SZ_WQKV + 255) / 256), 256, 0, stream>>>(
            Wqkv_f32[l], wqkv16[l], (int)SZ_WQKV);
        cvt_f32_f16<<<(int)((SZ_WO + 255) / 256), 256, 0, stream>>>(
            Wo_f32[l], wo16[l], (int)SZ_WO);
    }

    // ---- build sequences: cls + scaled embeddings + masks ----
    init_kernel<<<dim3(64, NSEQ), 256, 0, stream>>>(cls, seqbuf, maskbuf);
    aw_kernel<<<Bn * Nn, 64, 0, stream>>>(x, aw_W0, aw_b0, aw_W1, aw_b1,
                                          aw_W2, aw_b2, seqbuf, maskbuf);

    // z = ln(seq)
    ln_kernel<<<dim3(Spad, NSEQ), 256, 0, stream>>>(seqbuf, nullptr,
                                                    ln_g, ln_b, zbuf);

    for (int l = 0; l < 2; ++l) {
        const _Float16* ain = (l == 0) ? zbuf : abuf;
        gemm_wmma<true><<<dim3(MT64, 1536 / 64, NSEQ), 32, 0, stream>>>(
            ain, Dm, (long)Spad * Dm, wqkv16[l], bqkv[l],
            (void*)qkvbuf, 1536, (long)Spad * 1536, Dm);
        attn_wmma<<<dim3(MT, Hh, NSEQ), 32, 0, stream>>>(qkvbuf, maskbuf, obuf);
        gemm_wmma<false><<<dim3(MT64, Dm / 64, NSEQ), 32, 0, stream>>>(
            obuf, Dm, (long)Spad * Dm, wo16[l], bo[l],
            (void*)mhabuf, Dm, (long)Spad * Dm, Dm);
        ln_kernel<<<dim3(Spad, NSEQ), 256, 0, stream>>>(zbuf, mhabuf,
                                                        ln_g, ln_b, abuf);
    }

    proj_kernel<<<Bn, 256, 0, stream>>>(abuf, pr_W0, pr_b0, pr_W1, pr_b1,
                                        pr_W2, pr_b2, out);
    (void)in_sizes; (void)n_in; (void)out_size; (void)ws_size;
}
